// SO2_Attention_49529562858340
// MI455X (gfx1250) — matrix operand order repaired and checked by reading, not compiled, not run on hardware
//
#include <hip/hip_runtime.h>
#include <math.h>

typedef _Float16 v16h __attribute__((ext_vector_type(16)));
typedef _Float16 v8h  __attribute__((ext_vector_type(8)));
typedef float    v8f  __attribute__((ext_vector_type(8)));

#define N_NODES 10000
#define N_EDGES_C 160000
#define FEAT 240
#define LATENT 128
#define S3C 0.8660254037844386f

// ---------------------------------------------------------------------------
// Weight prep: src (K,N) f32 row-major  ->  dst (N,Kp) f16 row-major
// (transposed + zero-padded along K so every GEMM K is a multiple of 32).
// B-fragment for v_wmma_f32_16x16x32_f16 is then a contiguous 32B load:
// lane n (0..15) holds K = 0..15 of column n, lane n+16 holds K = 16..31.
// ---------------------------------------------------------------------------
__global__ void wconv_kernel(const float* __restrict__ src, _Float16* __restrict__ dst,
                             int K, int N, int Kp) {
  int idx = blockIdx.x * blockDim.x + threadIdx.x;
  if (idx >= N * Kp) return;
  int n = idx / Kp, kp = idx - n * Kp;
  float v = (kp < K) ? src[(size_t)kp * N + n] : 0.0f;
  dst[idx] = (_Float16)v;
}

// ---------------------------------------------------------------------------
// Per-node e3nn block-diagonal linears (center + neighbor), f32 in/out.
// ~169M MACs total -> negligible; keep it simple VALU.
// ---------------------------------------------------------------------------
__global__ void __launch_bounds__(256) node_linear_kernel(
    const float* __restrict__ x,
    const float* __restrict__ Wc0, const float* __restrict__ Wc1,
    const float* __restrict__ Wc2, const float* __restrict__ bc0,
    const float* __restrict__ Wn0, const float* __restrict__ Wn1,
    const float* __restrict__ Wn2, const float* __restrict__ bn0,
    float* __restrict__ outC, float* __restrict__ outN)
{
  __shared__ float xs[FEAT];
  const int node = blockIdx.x;
  const int t = threadIdx.x;
  if (t < FEAT) xs[t] = x[(size_t)node * FEAT + t];
  __syncthreads();
  if (t >= FEAT) return;

  int seg, k, q;
  if (t < 64)       { seg = 0; k = t;            q = 0;         }
  else if (t < 160) { seg = 1; int l = t - 64;  k = l / 3; q = l - 3 * k; }
  else              { seg = 2; int l = t - 160; k = l / 5; q = l - 5 * k; }

  #pragma unroll
  for (int which = 0; which < 2; ++which) {
    const float* W0 = which ? Wn0 : Wc0;
    const float* W1 = which ? Wn1 : Wc1;
    const float* W2 = which ? Wn2 : Wc2;
    const float* b0 = which ? bn0 : bc0;
    float acc = 0.0f;
    if (seg == 0) {
      for (int m = 0; m < 64; ++m) acc += xs[m] * W0[m * 64 + k];
      acc = acc * 0.125f + b0[k];                       // 1/sqrt(64)
    } else if (seg == 1) {
      for (int m = 0; m < 32; ++m) acc += xs[64 + m * 3 + q] * W1[m * 32 + k];
      acc *= 0.17677669529663687f;                      // 1/sqrt(32)
    } else {
      for (int m = 0; m < 16; ++m) acc += xs[160 + m * 5 + q] * W2[m * 16 + k];
      acc *= 0.25f;                                     // 1/sqrt(16)
    }
    float* o = which ? outN : outC;
    o[(size_t)node * FEAT + t] = acc;
  }
}

// ---------------------------------------------------------------------------
// Fused edge kernel: one wave32 handles 32 edges (2 WMMA M-tiles).
//   Phase 1 (lane==edge): rotation build + gather + rotate + product -> sim LDS
//   Phase 2: WMMA GEMM chain sim(240)->h(384,silu)->h1(256,silu)->out(128)
//   B-fragments cached in registers per N-tile and reused across both M-tiles
//   (halves L2 weight traffic, the dominant byte stream).
// LDS: sim 32x264 f16 (h1 reuses it) + h 32x392 f16 (f32 out-staging reuses it).
// ---------------------------------------------------------------------------
__device__ __constant__ float cJ5[25] = {
  0.f, 0.f, 0.f,  1.f, 0.f,
  0.f, 1.f, 0.f,  0.f, 0.f,
  0.f, 0.f,-0.5f, 0.f,-S3C,
  1.f, 0.f, 0.f,  0.f, 0.f,
  0.f, 0.f,-S3C,  0.f, 0.5f };

#define SIM_STR 264   // 256 used cols + pad (breaks LDS bank periodicity)
#define H_STR   392   // 384 used cols + pad

// A-fragment (16x32 f16): lane m holds K = hi*16..hi*16+15
__device__ __forceinline__ v16h ldsA(const _Float16* rowp /* row base + hi*16 */, int kbase) {
  v8h lo = *(const v8h*)(rowp + kbase);
  v8h hh = *(const v8h*)(rowp + kbase + 8);
  return __builtin_shufflevector(lo, hh, 0,1,2,3,4,5,6,7,8,9,10,11,12,13,14,15);
}

__device__ __forceinline__ v8f wmma_f16(v16h a, v16h b, v8f c) {
  return __builtin_amdgcn_wmma_f32_16x16x32_f16(false, a, false, b, (short)0, c, false, false);
}

// fast silu: f16-output precision doesn't need IEEE divide; one v_rcp_f32
__device__ __forceinline__ float silu(float x) {
  return x * __builtin_amdgcn_rcpf(1.0f + __expf(-x));
}

__global__ void __launch_bounds__(32) edge_kernel(
    const float* __restrict__ cF, const float* __restrict__ nF,
    const float* __restrict__ evec, const int* __restrict__ eidx,
    const _Float16* __restrict__ Wt0, const _Float16* __restrict__ Wt1,
    const _Float16* __restrict__ Wt2,
    const float* __restrict__ simb0, const float* __restrict__ simb1,
    const float* __restrict__ simb2,
    const _Float16* __restrict__ Wm1, const float* __restrict__ mlpb1,
    const _Float16* __restrict__ Wm2, const float* __restrict__ mlpb2,
    float* __restrict__ out, int nEdges)
{
  __shared__ __align__(32) _Float16 smem[32 * SIM_STR + 32 * H_STR];
  _Float16* sim  = smem;                  // 32 x SIM_STR   (also reused as h1)
  _Float16* hbuf = smem + 32 * SIM_STR;   // 32 x H_STR     (also reused as f32 out-staging)
  _Float16* h1   = smem;                  // alias of sim region

  const int t = threadIdx.x;              // 0..31, one wave
  const int blockBase = blockIdx.x * 32;
  const int e = blockBase + t;
  _Float16* srow = sim + t * SIM_STR;

  // ---------------- Phase 1: per-edge rotation + gather + product ----------
  if (e < nEdges) {
    const int src = eidx[e];
    const int dst = eidx[nEdges + e];
    const float* cp  = cF + (size_t)src * FEAT;
    const float* np_ = nF + (size_t)dst * FEAT;

    const float ex = evec[3 * e], ey = evec[3 * e + 1], ez = evec[3 * e + 2];
    float v0 = ey, v1 = ez, v2 = ex;                       // permutation [1,2,0]
    const float inv = rsqrtf(v0 * v0 + v1 * v1 + v2 * v2);
    v0 *= inv; v1 *= inv; v2 *= inv;
    const float alpha = atan2f(v0, v2);
    const float beta  = acosf(fminf(fmaxf(v1, -1.0f), 1.0f));
    const float sa  = __sinf(alpha),        ca  = __cosf(alpha);
    const float sb  = __sinf(beta),         cb  = __cosf(beta);
    const float s2a = __sinf(2.0f * alpha), c2a = __cosf(2.0f * alpha);
    const float s2b = __sinf(2.0f * beta),  c2b = __cosf(2.0f * beta);

    // seg0 (mul=64, d=1): plain elementwise product
    #pragma unroll
    for (int k = 0; k < 64; k += 4) {
      float4 c4 = *(const float4*)(cp + k);
      float4 n4 = *(const float4*)(np_ + k);
      srow[k + 0] = (_Float16)(c4.x * n4.x);
      srow[k + 1] = (_Float16)(c4.y * n4.y);
      srow[k + 2] = (_Float16)(c4.z * n4.z);
      srow[k + 3] = (_Float16)(c4.w * n4.w);
    }

    // seg1 (mul=32, d=3): R3 = Za3 @ J1 @ Zb3 @ J1 (closed form)
    const float R3[3][3] = {
      { ca,  sa * sb,  sa * cb },
      { 0.f, cb,      -sb      },
      { -sa, ca * sb,  ca * cb } };
    for (int m = 0; m < 32; ++m) {
      const float* c3 = cp  + 64 + m * 3;
      const float* n3 = np_ + 64 + m * 3;
      #pragma unroll
      for (int q = 0; q < 3; ++q) {
        float cr = c3[0] * R3[0][q] + c3[1] * R3[1][q] + c3[2] * R3[2][q];
        float nr = n3[0] * R3[0][q] + n3[1] * R3[1][q] + n3[2] * R3[2][q];
        srow[64 + m * 3 + q] = (_Float16)(cr * nr);
      }
    }

    // seg2 (mul=16, d=5): R5 = Za5 @ J2 @ Zb5 @ J2 via sparse Z structure
    float Y[5][5], M5[5][5], R5[5][5];
    #pragma unroll
    for (int j = 0; j < 5; ++j) {
      Y[0][j] = c2b * cJ5[0 * 5 + j] + s2b * cJ5[4 * 5 + j];
      Y[1][j] = cb  * cJ5[1 * 5 + j] + sb  * cJ5[3 * 5 + j];
      Y[2][j] = cJ5[2 * 5 + j];
      Y[3][j] = -sb  * cJ5[1 * 5 + j] + cb  * cJ5[3 * 5 + j];
      Y[4][j] = -s2b * cJ5[0 * 5 + j] + c2b * cJ5[4 * 5 + j];
    }
    #pragma unroll
    for (int j = 0; j < 5; ++j) {            // M5 = J2 @ Y (J2 rows are sparse)
      M5[0][j] = Y[3][j];
      M5[1][j] = Y[1][j];
      M5[2][j] = -0.5f * Y[2][j] - S3C * Y[4][j];
      M5[3][j] = Y[0][j];
      M5[4][j] = -S3C * Y[2][j] + 0.5f * Y[4][j];
    }
    #pragma unroll
    for (int j = 0; j < 5; ++j) {            // R5 = Za5 @ M5
      R5[0][j] = c2a * M5[0][j] + s2a * M5[4][j];
      R5[1][j] = ca  * M5[1][j] + sa  * M5[3][j];
      R5[2][j] = M5[2][j];
      R5[3][j] = -sa  * M5[1][j] + ca  * M5[3][j];
      R5[4][j] = -s2a * M5[0][j] + c2a * M5[4][j];
    }
    for (int m = 0; m < 16; ++m) {
      const float* c5 = cp  + 160 + m * 5;
      const float* n5 = np_ + 160 + m * 5;
      #pragma unroll
      for (int q = 0; q < 5; ++q) {
        float cr = c5[0]*R5[0][q] + c5[1]*R5[1][q] + c5[2]*R5[2][q] + c5[3]*R5[3][q] + c5[4]*R5[4][q];
        float nr = n5[0]*R5[0][q] + n5[1]*R5[1][q] + n5[2]*R5[2][q] + n5[3]*R5[3][q] + n5[4]*R5[4][q];
        srow[160 + m * 5 + q] = (_Float16)(cr * nr);
      }
    }
    // zero-pad seg2 K 80 -> 96
    #pragma unroll
    for (int k = 240; k < 256; ++k) srow[k] = (_Float16)0.0f;
  } else {
    for (int k = 0; k < 256; ++k) srow[k] = (_Float16)0.0f;
  }
  __syncthreads();

  // ---------------- Phase 2: WMMA GEMM chain (wave-local tiles) ------------
  const int ln = t & 15;     // A row (M) / B+C column (N) index within tile
  const int hi = t >> 4;     // half-wave selector
  const int segBase[3] = { 0, 64, 160 };
  const int segK[3]    = { 64, 96, 96 };
  const _Float16* Wts[3]   = { Wt0, Wt1, Wt2 };
  const float*    sbias[3] = { simb0, simb1, simb2 };

  // --- stage A: sim -> h (3 block GEMMs, K=64/96/96, N=128 each) ---
  #pragma unroll
  for (int seg = 0; seg < 3; ++seg) {
    const int nkc = segK[seg] >> 5;
    const int Kp  = segK[seg];
    for (int nt = 0; nt < 8; ++nt) {
      v16h bfr[3];                                   // B cached across both M-tiles
      const _Float16* wp = Wts[seg] + (size_t)(nt * 16 + ln) * Kp + hi * 16;
      #pragma unroll
      for (int kc = 0; kc < nkc; ++kc) bfr[kc] = *(const v16h*)(wp + kc * 32);
      const float bv = sbias[seg][nt * 16 + ln];
      #pragma unroll
      for (int mt = 0; mt < 2; ++mt) {
        const _Float16* arow = sim + (mt * 16 + ln) * SIM_STR + hi * 16;
        v8f acc = {};
        #pragma unroll
        for (int kc = 0; kc < nkc; ++kc)
          acc = wmma_f16(ldsA(arow, segBase[seg] + kc * 32), bfr[kc], acc);
        _Float16* hp = hbuf + (mt * 16 + hi * 8) * H_STR + seg * 128 + nt * 16 + ln;
        #pragma unroll
        for (int r = 0; r < 8; ++r)
          hp[r * H_STR] = (_Float16)silu(acc[r] + bv);   // row = mt*16 + hi*8 + r
      }
    }
  }
  __syncthreads();

  // --- stage B: h (384) -> h1 (256), silu; h1 overwrites sim region ---
  for (int nt = 0; nt < 16; ++nt) {
    v16h bfr[12];
    const _Float16* wp = Wm1 + (size_t)(nt * 16 + ln) * 384 + hi * 16;
    #pragma unroll
    for (int kc = 0; kc < 12; ++kc) bfr[kc] = *(const v16h*)(wp + kc * 32);
    const float bv = mlpb1[nt * 16 + ln];
    #pragma unroll
    for (int mt = 0; mt < 2; ++mt) {
      const _Float16* arow = hbuf + (mt * 16 + ln) * H_STR + hi * 16;
      v8f acc = {};
      #pragma unroll
      for (int kc = 0; kc < 12; ++kc)
        acc = wmma_f16(ldsA(arow, kc * 32), bfr[kc], acc);
      _Float16* hp = h1 + (mt * 16 + hi * 8) * SIM_STR + nt * 16 + ln;
      #pragma unroll
      for (int r = 0; r < 8; ++r)
        hp[r * SIM_STR] = (_Float16)silu(acc[r] + bv);
    }
  }
  __syncthreads();

  // --- stage C: h1 (256) -> out (128); stage f32 results in LDS (hbuf is dead) ---
  float* stg = (float*)hbuf;                 // 32 x 128 f32 = 16KB, fits in hbuf region
  for (int nt = 0; nt < 8; ++nt) {
    v16h bfr[8];
    const _Float16* wp = Wm2 + (size_t)(nt * 16 + ln) * 256 + hi * 16;
    #pragma unroll
    for (int kc = 0; kc < 8; ++kc) bfr[kc] = *(const v16h*)(wp + kc * 32);
    const float bv = mlpb2[nt * 16 + ln];
    #pragma unroll
    for (int mt = 0; mt < 2; ++mt) {
      const _Float16* arow = h1 + (mt * 16 + ln) * SIM_STR + hi * 16;
      v8f acc = {};
      #pragma unroll
      for (int kc = 0; kc < 8; ++kc)
        acc = wmma_f16(ldsA(arow, kc * 32), bfr[kc], acc);
      #pragma unroll
      for (int r = 0; r < 8; ++r)
        stg[(mt * 16 + hi * 8 + r) * 128 + nt * 16 + ln] = acc[r] + bv;
    }
  }
  __syncthreads();

  // coalesced output: one 512B (32-lane float4) store per row
  const float4* stg4 = (const float4*)stg;
  float4* out4 = (float4*)out;
  for (int row = 0; row < 32; ++row) {
    const int edge = blockBase + row;
    if (edge < nEdges)
      out4[(size_t)edge * (LATENT / 4) + t] = stg4[row * 32 + t];
  }
}

// ---------------------------------------------------------------------------
// Host launcher
// ---------------------------------------------------------------------------
extern "C" void kernel_launch(void* const* d_in, const int* in_sizes, int n_in,
                              void* d_out, int out_size, void* d_ws, size_t ws_size,
                              hipStream_t stream) {
  (void)in_sizes; (void)n_in; (void)out_size; (void)ws_size;

  const float* node_features = (const float*)d_in[0];
  const float* evec = (const float*)d_in[1];
  const int*   eidx = (const int*)d_in[2];
  // params in dict insertion order
  const float* Wc0 = (const float*)d_in[3];
  const float* Wc1 = (const float*)d_in[4];
  const float* Wc2 = (const float*)d_in[5];
  const float* bc0 = (const float*)d_in[6];
  const float* Wn0 = (const float*)d_in[7];
  const float* Wn1 = (const float*)d_in[8];
  const float* Wn2 = (const float*)d_in[9];
  const float* bn0 = (const float*)d_in[10];
  const float* simW0 = (const float*)d_in[11];
  const float* simb0 = (const float*)d_in[12];
  const float* simW1 = (const float*)d_in[13];
  const float* simb1 = (const float*)d_in[14];
  const float* simW2 = (const float*)d_in[15];
  const float* simb2 = (const float*)d_in[16];
  const float* mlpW1 = (const float*)d_in[17];
  const float* mlpb1 = (const float*)d_in[18];
  const float* mlpW2 = (const float*)d_in[19];
  const float* mlpb2 = (const float*)d_in[20];
  float* outp = (float*)d_out;

  // workspace layout (all offsets 32B-aligned)
  char* ws = (char*)d_ws;
  float* cF = (float*)ws;                                   //  9,600,000 B
  float* nF = cF + (size_t)N_NODES * FEAT;                  //  9,600,000 B
  _Float16* Wt0  = (_Float16*)(ws + (size_t)2 * N_NODES * FEAT * sizeof(float));
  _Float16* Wt1  = Wt0  + 128 * 64;                         // (128,64)
  _Float16* Wt2  = Wt1  + 128 * 96;                         // (128,96)
  _Float16* Wm1t = Wt2  + 128 * 96;                         // (128,96) padded
  _Float16* Wm2t = Wm1t + 256 * 384;                        // (256,384)

  // 1) weight transpose/convert to f16 (padded K)
  {
    int tot;
    tot = 128 * 64;  wconv_kernel<<<(tot + 255) / 256, 256, 0, stream>>>(simW0, Wt0, 64, 128, 64);
    tot = 128 * 96;  wconv_kernel<<<(tot + 255) / 256, 256, 0, stream>>>(simW1, Wt1, 96, 128, 96);
    tot = 128 * 96;  wconv_kernel<<<(tot + 255) / 256, 256, 0, stream>>>(simW2, Wt2, 80, 128, 96);
    tot = 256 * 384; wconv_kernel<<<(tot + 255) / 256, 256, 0, stream>>>(mlpW1, Wm1t, 384, 256, 384);
    tot = 128 * 256; wconv_kernel<<<(tot + 255) / 256, 256, 0, stream>>>(mlpW2, Wm2t, 256, 128, 256);
  }

  // 2) per-node center/neighbor linears
  node_linear_kernel<<<N_NODES, 256, 0, stream>>>(
      node_features, Wc0, Wc1, Wc2, bc0, Wn0, Wn1, Wn2, bn0, cF, nF);

  // 3) fused per-edge kernel: 32 edges per wave32 workgroup
  const int nBlocks = (N_EDGES_C + 31) / 32;
  edge_kernel<<<nBlocks, 32, 0, stream>>>(
      cF, nF, evec, eidx, Wt0, Wt1, Wt2, simb0, simb1, simb2,
      Wm1t, mlpb1, Wm2t, mlpb2, outp, N_EDGES_C);
}
